// RNNModel_36653250904452
// MI455X (gfx1250) — compile-verified, matrix operand order checked by reading, not run on hardware
//
#include <hip/hip_runtime.h>
#include <hip/hip_bf16.h>
#include <math.h>

typedef __bf16 bf16_t;
typedef __attribute__((ext_vector_type(16))) __bf16 v16bf;
typedef __attribute__((ext_vector_type(8)))  __bf16 v8bf;
typedef __attribute__((ext_vector_type(8)))  float  v8f;

#define S_LEN 1024
#define BATCH 256
#define EDIM  128
#define HDIM  256
#define VOCAB 32000
#define KDIM  384      // E + H
#define NKB   12       // K blocks of 32
#define NTILES 64      // 4H / 16 column tiles
#define ROWS  16
#define COLS  32
#define SASTR 392      // padded LDS row stride in bf16 (784B: 16B aligned, bank-skewed)

// workspace layout (bytes)
#define WSW_OFF    0
#define WSW_BYTES  (NTILES * NKB * 32 * 16 * 2)     // 786432
#define HBUF_BYTES (BATCH * HDIM * 2)               // 131072
#define HBUF0_OFF  (WSW_OFF + WSW_BYTES)
#define HBUF1_OFF  (HBUF0_OFF + HBUF_BYTES)
#define CBUF_OFF   (HBUF1_OFF + HBUF_BYTES)
#define CBUF_BYTES (BATCH * HDIM * 4)
#define EMB_OFF    (CBUF_OFF + CBUF_BYTES)          // bf16 embedding table, 8.192 MB

// ---------------------------------------------------------------------------
// One-time: convert embedding table f32 -> bf16 (packed as dwords).
// ---------------------------------------------------------------------------
__global__ void lstm_cvt_emb(const float* __restrict__ emb,
                             unsigned int* __restrict__ embbf32) {
    int i = blockIdx.x * blockDim.x + threadIdx.x;   // one per bf16 pair
    if (i >= VOCAB * EDIM / 2) return;
    bf16_t a = (bf16_t)emb[2 * i];
    bf16_t b = (bf16_t)emb[2 * i + 1];
    unsigned short ua = __builtin_bit_cast(unsigned short, a);
    unsigned short ub = __builtin_bit_cast(unsigned short, b);
    embbf32[i] = (unsigned int)ua | ((unsigned int)ub << 16);
}

// ---------------------------------------------------------------------------
// One-time weight swizzle: Wcat[k][n] = (k<E ? w_ih[n][k] : w_hh[n][k-E]),
// stored in exact WMMA B-fragment per-lane layout:
//   chunk(nt, kb): 32 lanes x 16 bf16, lane L holds col n = nt*16 + (L&15),
//   element i holds k = kb*32 + i + (i<8?0:8) + (L<16?0:8)
// ---------------------------------------------------------------------------
__global__ void lstm_swizzle_weights(const float* __restrict__ w_ih,
                                     const float* __restrict__ w_hh,
                                     bf16_t* __restrict__ wsw) {
    int gid = blockIdx.x * blockDim.x + threadIdx.x;   // one per (nt, kb, lane)
    if (gid >= NTILES * NKB * 32) return;
    int lane = gid & 31;
    int kb   = (gid >> 5) % NKB;
    int nt   = gid / (NKB * 32);
    int n    = nt * 16 + (lane & 15);
    int kbase = kb * 32 + ((lane < 16) ? 0 : 8);
    bf16_t* dst = wsw + gid * 16;
#pragma unroll
    for (int i = 0; i < 16; ++i) {
        int k = kbase + i + ((i < 8) ? 0 : 8);
        float v = (k < EDIM) ? w_ih[n * EDIM + k] : w_hh[n * HDIM + (k - EDIM)];
        dst[i] = (bf16_t)v;
    }
}

__global__ void lstm_init_state(const float* __restrict__ h0,
                                const float* __restrict__ c0,
                                bf16_t* __restrict__ hbuf0,
                                float* __restrict__ cbuf) {
    int i = blockIdx.x * blockDim.x + threadIdx.x;
    if (i < BATCH * HDIM) {
        hbuf0[i] = (bf16_t)h0[i];
        cbuf[i]  = c0[i];
    }
}

// ---------------------------------------------------------------------------
// One LSTM time step. Grid: 128 WGs (16 row groups x 8 col groups of 32).
// Block: 128 threads = 4 waves; wave w computes gate w over [16 x 32] cols.
// A tile staged to LDS with async global->LDS B128 copies (ASYNCcnt path).
// ---------------------------------------------------------------------------
__launch_bounds__(128)
__global__ void lstm_step(int s,
                          const int*    __restrict__ inp,
                          const int*    __restrict__ lengths,
                          const bf16_t* __restrict__ embbf,
                          const bf16_t* __restrict__ wsw,
                          const float*  __restrict__ b_ih,
                          const float*  __restrict__ b_hh,
                          const bf16_t* __restrict__ h_in,
                          bf16_t*       __restrict__ h_out,
                          float*        __restrict__ c_st,
                          float*        __restrict__ last_h,
                          float*        __restrict__ last_c) {
    __shared__ __align__(16) bf16_t sA[ROWS * SASTR];       // [16 rows][K=384] bf16
    __shared__ float sG[4 * ROWS * COLS];                   // gate results f32
    __shared__ int   sTok[ROWS];

    const int tid = threadIdx.x;
    const int r0  = (blockIdx.x >> 3) * ROWS;               // batch row base
    const int c0  = (blockIdx.x & 7) * COLS;                // col base within a gate

    if (tid < ROWS) sTok[tid] = inp[s * BATCH + r0 + tid];
    __syncthreads();

    const unsigned sAbase = (unsigned)(uintptr_t)(&sA[0]);  // LDS offset (low 32b)

    // Stage A (x part): 16 rows x 256B, 256 x 16B chunks, async direct-to-LDS
#pragma unroll
    for (int it = 0; it < 2; ++it) {
        int ch = tid + it * 128;            // 0..255
        int r  = ch >> 4, co = ch & 15;
        const bf16_t* g = embbf + ((long)sTok[r] * EDIM + co * 8);
        unsigned la = sAbase + (unsigned)(r * SASTR * 2 + co * 16);
        asm volatile("global_load_async_to_lds_b128 %0, %1, off"
                     :: "v"(la), "v"(g) : "memory");
    }
    // Stage A (h part): 16 rows x 512B, 512 x 16B chunks, async direct-to-LDS
#pragma unroll
    for (int it = 0; it < 4; ++it) {
        int ch = tid + it * 128;            // 0..511
        int r  = ch >> 5, co = ch & 31;
        const bf16_t* g = h_in + ((r0 + r) * HDIM + co * 8);
        unsigned la = sAbase + (unsigned)(r * SASTR * 2 + 256 + co * 16);
        asm volatile("global_load_async_to_lds_b128 %0, %1, off"
                     :: "v"(la), "v"(g) : "memory");
    }
    asm volatile("s_wait_asynccnt 0x0" ::: "memory");
    __syncthreads();

    const int wv   = tid >> 5;          // gate index 0..3 (i,f,g,o)
    const int lane = tid & 31;
    const int mrow = lane & 15;
    const int hi   = (lane < 16) ? 0 : 8;
    const int nt0  = wv * 16 + (c0 >> 4);   // global column tile of 16

    v8f acc0 = {}; v8f acc1 = {};
#pragma unroll
    for (int kb = 0; kb < NKB; ++kb) {
        // A fragment: lane holds row M=lane%16, K = kb*32 + {0..7,16..23} (+8 for hi half)
        const bf16_t* ap = &sA[mrow * SASTR + kb * 32 + hi];
        v8bf alo = *(const v8bf*)(ap);
        v8bf ahi = *(const v8bf*)(ap + 16);
        v16bf a = __builtin_shufflevector(alo, ahi,
                    0,1,2,3,4,5,6,7,8,9,10,11,12,13,14,15);
        {   // tile 0
            const bf16_t* bp = &wsw[(((nt0 + 0) * NKB + kb) * 32 + lane) * 16];
            v8bf blo = *(const v8bf*)(bp);
            v8bf bhi = *(const v8bf*)(bp + 8);
            v16bf b = __builtin_shufflevector(blo, bhi,
                        0,1,2,3,4,5,6,7,8,9,10,11,12,13,14,15);
            acc0 = __builtin_amdgcn_wmma_f32_16x16x32_bf16(
                       false, a, false, b, (short)0, acc0, false, false);
        }
        {   // tile 1
            const bf16_t* bp = &wsw[(((nt0 + 1) * NKB + kb) * 32 + lane) * 16];
            v8bf blo = *(const v8bf*)(bp);
            v8bf bhi = *(const v8bf*)(bp + 8);
            v16bf b = __builtin_shufflevector(blo, bhi,
                        0,1,2,3,4,5,6,7,8,9,10,11,12,13,14,15);
            acc1 = __builtin_amdgcn_wmma_f32_16x16x32_bf16(
                       false, a, false, b, (short)0, acc1, false, false);
        }
    }

    // C/D layout: lanes 0-15 -> N=lane, VGPR r -> M=r; lanes 16-31 -> M=r+8
#pragma unroll
    for (int r = 0; r < 8; ++r) {
        int M = r + hi;
        sG[(wv * ROWS + M) * COLS + (lane & 15)]      = acc0[r];
        sG[(wv * ROWS + M) * COLS + 16 + (lane & 15)] = acc1[r];
    }
    __syncthreads();

    // Fused LSTM cell update: 16x32 = 512 elements over 128 threads
#pragma unroll
    for (int it = 0; it < 4; ++it) {
        int e  = tid + it * 128;
        int r  = e >> 5;
        int jl = e & 31;
        int bb = r0 + r;
        int j  = c0 + jl;
        float gi = sG[(0 * ROWS + r) * COLS + jl] + b_ih[0 * HDIM + j] + b_hh[0 * HDIM + j];
        float gf = sG[(1 * ROWS + r) * COLS + jl] + b_ih[1 * HDIM + j] + b_hh[1 * HDIM + j];
        float gg = sG[(2 * ROWS + r) * COLS + jl] + b_ih[2 * HDIM + j] + b_hh[2 * HDIM + j];
        float go = sG[(3 * ROWS + r) * COLS + jl] + b_ih[3 * HDIM + j] + b_hh[3 * HDIM + j];
        float i_ = 1.0f / (1.0f + expf(-gi));
        float f_ = 1.0f / (1.0f + expf(-gf));
        float o_ = 1.0f / (1.0f + expf(-go));
        float g_ = tanhf(gg);
        float cp = c_st[bb * HDIM + j];
        float cn = f_ * cp + i_ * g_;
        float hn = o_ * tanhf(cn);
        c_st[bb * HDIM + j]  = cn;
        h_out[bb * HDIM + j] = (bf16_t)hn;
        if (s == lengths[bb] - 1) {
            last_h[bb * HDIM + j] = hn;
            last_c[bb * HDIM + j] = cn;
        }
    }
}

__global__ void lstm_decode(const float* __restrict__ last_h,
                            const float* __restrict__ dec_w,
                            float* __restrict__ decoded) {
    int b = blockIdx.x * blockDim.x + threadIdx.x;
    if (b >= BATCH) return;
#pragma unroll
    for (int o = 0; o < 2; ++o) {
        float acc = 0.0f;
        for (int k = 0; k < HDIM; ++k)
            acc += last_h[b * HDIM + k] * dec_w[o * HDIM + k];
        decoded[b * 2 + o] = 1.0f / (1.0f + expf(-acc));
    }
}

extern "C" void kernel_launch(void* const* d_in, const int* in_sizes, int n_in,
                              void* d_out, int out_size, void* d_ws, size_t ws_size,
                              hipStream_t stream) {
    (void)in_sizes; (void)n_in; (void)out_size; (void)ws_size;
    const int*   inp     = (const int*)d_in[0];
    const int*   lengths = (const int*)d_in[1];
    const float* h0      = (const float*)d_in[2];
    const float* c0      = (const float*)d_in[3];
    const float* emb     = (const float*)d_in[4];
    const float* w_ih    = (const float*)d_in[5];
    const float* w_hh    = (const float*)d_in[6];
    const float* b_ih    = (const float*)d_in[7];
    const float* b_hh    = (const float*)d_in[8];
    const float* dec_w   = (const float*)d_in[9];

    char*   ws    = (char*)d_ws;
    bf16_t* wsw   = (bf16_t*)(ws + WSW_OFF);
    bf16_t* hb0   = (bf16_t*)(ws + HBUF0_OFF);
    bf16_t* hb1   = (bf16_t*)(ws + HBUF1_OFF);
    float*  cbuf  = (float*)(ws + CBUF_OFF);
    bf16_t* embbf = (bf16_t*)(ws + EMB_OFF);

    float* decoded = (float*)d_out;
    float* last_h  = decoded + BATCH * 2;
    float* last_c  = last_h + BATCH * HDIM;

    lstm_cvt_emb<<<(VOCAB * EDIM / 2 + 255) / 256, 256, 0, stream>>>(
        emb, (unsigned int*)embbf);
    lstm_swizzle_weights<<<(NTILES * NKB * 32 + 255) / 256, 256, 0, stream>>>(w_ih, w_hh, wsw);
    lstm_init_state<<<(BATCH * HDIM + 255) / 256, 256, 0, stream>>>(h0, c0, hb0, cbuf);

    bf16_t* hbuf[2] = { hb0, hb1 };
    for (int s = 0; s < S_LEN; ++s) {
        lstm_step<<<128, 128, 0, stream>>>(s, inp, lengths, embbf, wsw, b_ih, b_hh,
                                           hbuf[s & 1], hbuf[(s + 1) & 1], cbuf,
                                           last_h, last_c);
    }
    lstm_decode<<<1, 256, 0, stream>>>(last_h, dec_w, decoded);
}
